// EnsembleRSSM_15255723835609
// MI455X (gfx1250) — matrix operand order repaired and checked by reading, not compiled, not run on hardware
//
#include <hip/hip_runtime.h>
#include <hip/hip_bf16.h>
#include <math.h>

// ---------------- problem constants ----------------
#define B_      256
#define T_      64
#define STOCH_  32
#define DETER_  1024
#define HIDDEN_ 1024
#define EMBED_  1536
#define ACT_    6
#define ENS_    5
#define OUTW_   1216         // 6*STOCH + DETER per (b,t)
#define LN_EPS_ 1e-5f

typedef __attribute__((ext_vector_type(16))) _Float16 v16h;
typedef __attribute__((ext_vector_type(8)))  _Float16 v8h;
typedef __attribute__((ext_vector_type(8)))  float    v8f;

__device__ __forceinline__ float elu_f(float x)      { return x > 0.f ? x : expm1f(x); }
__device__ __forceinline__ float softplus_f(float x) { return x > 20.f ? x : log1pf(expf(x)); }
__device__ __forceinline__ float sigmoid_f(float x)  { return 1.f / (1.f + expf(-x)); }

// CDNA5 async memory->LDS copy (16B per lane), tracked by ASYNCcnt.
__device__ __forceinline__ void async_to_lds_b128(unsigned lds_addr, const void* gaddr) {
    asm volatile("global_load_async_to_lds_b128 %0, %1, off"
                 :: "v"(lds_addr), "v"(gaddr) : "memory");
}

// ---------------------------------------------------------------------------
// Weight convert + transpose:  W[K][N] (f32) -> Wt[N][Kpad] (f16), zero pad K.
// ---------------------------------------------------------------------------
__global__ void rssm_conv_w(const float* __restrict__ W, _Float16* __restrict__ Wt,
                            int K, int N, int Kpad)
{
    long idx = (long)blockIdx.x * blockDim.x + threadIdx.x;
    long total = (long)N * Kpad;
    if (idx >= total) return;
    int n = (int)(idx / Kpad);
    int k = (int)(idx % Kpad);
    Wt[idx] = (k < K) ? (_Float16)W[(size_t)k * N + n] : (_Float16)0.f;
}

// ---------------------------------------------------------------------------
// Per-step prepare: mask carries on episode start, build f16 activations.
// ---------------------------------------------------------------------------
__global__ __launch_bounds__(256)
void rssm_prep(const float* __restrict__ embed, const float* __restrict__ action,
               const unsigned char* __restrict__ isFirst,
               const float* __restrict__ stochC, float* __restrict__ deterC,
               _Float16* __restrict__ Xinp, _Float16* __restrict__ Xgru,
               _Float16* __restrict__ Xobs, int t)
{
    const int b = blockIdx.x;
    const int tid = threadIdx.x;
    const float mf = isFirst[(size_t)b * T_ + t] ? 0.f : 1.f;

    for (int i = tid; i < 64; i += 256) {
        float v = 0.f;
        if (i < STOCH_)             v = (t == 0) ? 0.f : stochC[(size_t)b * STOCH_ + i] * mf;
        else if (i < STOCH_ + ACT_) v = action[((size_t)b * T_ + t) * ACT_ + (i - STOCH_)] * mf;
        Xinp[(size_t)b * 64 + i] = (_Float16)v;
    }
    for (int i = tid; i < DETER_; i += 256) {
        float d = (t == 0) ? 0.f : deterC[(size_t)b * DETER_ + i] * mf;
        deterC[(size_t)b * DETER_ + i] = d;
        Xgru[(size_t)b * 2048 + DETER_ + i] = (_Float16)d;
    }
    for (int i = tid; i < EMBED_; i += 256)
        Xobs[(size_t)b * 2560 + DETER_ + i] =
            (_Float16)embed[((size_t)b * T_ + t) * EMBED_ + i];
}

// ---------------------------------------------------------------------------
// WMMA GEMM:  out = epi( A[M=256][K](f16) @ Wt[N][K](f16)^T + bias )
// 64x64 block tile, BK=32, 128 threads = 4 waves, each wave 2x2 16x16 tiles.
// 4-buffer GLOBAL_LOAD_ASYNC_TO_LDS_B128 pipeline (ASYNCcnt, 4 ops per tile):
//   prologue: stage tiles 0..2 into buffers 0..2
//   steady j: s_wait_asynccnt 8  (tile j landed; j+1, j+2 in flight)
//             barrier            (buffer (j-1)&3 free of readers)
//             stage tile j+3 -> buffer (j+3)&3 == (j-1)&3
//             compute tile j from buffer j&3 ; pin accs before next wait
//   tail: peeled with literal s_wait_asynccnt 4 then 0 (no runtime selection).
// EPI: 0 = raw f32 (GRU pre-LN), 1 = elu -> f16, 2 = dist head -> d_out slots.
// ---------------------------------------------------------------------------
template<int EPI>
__global__ __launch_bounds__(128)
void rssm_gemm(const _Float16* __restrict__ A, int lda,
               const _Float16* __restrict__ Wt, int ldw,
               const float* __restrict__ bias, int K,
               float* __restrict__ outF32, _Float16* __restrict__ outF16, int ldo,
               const int* __restrict__ ensIdx, int t, long wSel, int bSel,
               float* __restrict__ dout, int meanOff, int stdOff)
{
    if (ensIdx) {               // ensemble head selection on device
        int e = ensIdx[t];
        Wt   += (long)e * wSel;
        bias += (long)e * bSel;
    }
    __shared__ alignas(16) _Float16 As[4][64 * 32];
    __shared__ alignas(16) _Float16 Bs[4][64 * 32];

    const int tid  = threadIdx.x;
    const int wave = tid >> 5, lane = tid & 31;
    const int wm = wave & 1, wn = wave >> 1;     // 2x2 wave grid
    const int l  = lane & 15, hi = lane >> 4;
    const int m0 = blockIdx.x * 64;
    const int n0 = blockIdx.y * 64;

    const int cr = tid >> 1;            // staging: row 0..63
    const int cc = (tid & 1) * 16;      // half-offset 0 / 16

    const _Float16* gA = A  + (size_t)(m0 + cr) * lda + cc;
    const _Float16* gB = Wt + (size_t)(n0 + cr) * ldw + cc;
    const unsigned bufStride = 64u * 32u * 2u;
    const unsigned ldsA0 = (unsigned)(size_t)(&As[0][0]) + (unsigned)(cr * 32 + cc) * 2u;
    const unsigned ldsB0 = (unsigned)(size_t)(&Bs[0][0]) + (unsigned)(cr * 32 + cc) * 2u;

    auto stage = [&](int buf, int kk) {      // 4 async b128 per thread per tile
        const _Float16* pa = gA + kk;
        const _Float16* pb = gB + kk;
        const unsigned la = ldsA0 + (unsigned)buf * bufStride;
        const unsigned lb = ldsB0 + (unsigned)buf * bufStride;
        async_to_lds_b128(la,       pa);
        async_to_lds_b128(la + 16u, pa + 8);
        async_to_lds_b128(lb,       pb);
        async_to_lds_b128(lb + 16u, pb + 8);
    };

    v8f acc[2][2] = {};

    auto compute = [&](int buf) {
        v16h a[2], b[2];
#pragma unroll
        for (int ii = 0; ii < 2; ++ii) {  // A frag: lanes 0-15: K0-7,16-23 ; 16-31: K8-15,24-31
            const _Float16* p = &As[buf][(wm * 32 + ii * 16 + l) * 32];
            v8h lo = *(const v8h*)(p + hi * 8);
            v8h hh = *(const v8h*)(p + 16 + hi * 8);
#pragma unroll
            for (int q = 0; q < 8; ++q) { a[ii][q] = lo[q]; a[ii][8 + q] = hh[q]; }
        }
#pragma unroll
        for (int jj = 0; jj < 2; ++jj) {  // B frag: lanes 0-15: K0-7,16-23 ; 16-31: K8-15,24-31
            const _Float16* p = &Bs[buf][(wn * 32 + jj * 16 + l) * 32];
            v8h lo = *(const v8h*)(p + hi * 8);
            v8h hh = *(const v8h*)(p + 16 + hi * 8);
#pragma unroll
            for (int q = 0; q < 8; ++q) { b[jj][q] = lo[q]; b[jj][8 + q] = hh[q]; }
        }
#pragma unroll
        for (int ii = 0; ii < 2; ++ii)
#pragma unroll
            for (int jj = 0; jj < 2; ++jj)
                acc[ii][jj] = __builtin_amdgcn_wmma_f32_16x16x32_f16(
                    false, a[ii], false, b[jj], (short)0, acc[ii][jj], false, false);

        // Pin accumulators: keeps the WMMAs scheduled before the next
        // s_wait_asynccnt (pure-register ops would otherwise sink past it).
        asm volatile("" : "+v"(acc[0][0]), "+v"(acc[0][1]),
                          "+v"(acc[1][0]), "+v"(acc[1][1]));
    };

    const int NT = K >> 5;                   // K / 32 tiles (always >= 2 here)
    stage(0, 0);
    if (NT > 1) stage(1, 32);
    if (NT > 2) stage(2, 64);

    int j = 0;
    for (; j < NT - 2; ++j) {                // steady state: 2-3 tiles in flight
        asm volatile("s_wait_asynccnt 8" ::: "memory");
        __syncthreads();
        if (j + 3 < NT) stage((j + 3) & 3, (j + 3) * 32);
        compute(j & 3);
    }
    asm volatile("s_wait_asynccnt 4" ::: "memory");   // tile NT-2 landed
    __syncthreads();
    compute(j & 3);
    ++j;
    asm volatile("s_wait_asynccnt 0" ::: "memory");   // tile NT-1 landed
    __syncthreads();
    compute(j & 3);

    // epilogue: C/D layout — VGPR r -> M = hi*8 + r, N = l
#pragma unroll
    for (int i = 0; i < 2; ++i)
#pragma unroll
        for (int jx = 0; jx < 2; ++jx)
#pragma unroll
            for (int r = 0; r < 8; ++r) {
                const int gm = m0 + wm * 32 + i * 16 + hi * 8 + r;
                const int gn = n0 + wn * 32 + jx * 16 + l;
                float v = acc[i][jx][r] + bias[gn];
                if constexpr (EPI == 0) {
                    outF32[(size_t)gm * ldo + gn] = v;
                } else if constexpr (EPI == 1) {
                    outF16[(size_t)gm * ldo + gn] = (_Float16)elu_f(v);
                } else {
                    const size_t base = ((size_t)gm * T_ + t) * OUTW_;
                    if (gn < STOCH_) dout[base + meanOff + gn] = v;
                    else             dout[base + stdOff + (gn - STOCH_)] =
                                         softplus_f(v) + 0.1f;
                }
            }
}

// ---------------------------------------------------------------------------
// GRU gate: per-row LayerNorm over 3072, gates, deter_new; one block per row.
// ---------------------------------------------------------------------------
__global__ __launch_bounds__(256)
void rssm_gate(const float* __restrict__ Y, const float* __restrict__ ln_g,
               const float* __restrict__ ln_b, float* __restrict__ deterC,
               _Float16* __restrict__ Xobs, _Float16* __restrict__ Xens,
               float* __restrict__ dout, int t)
{
    const int b = blockIdx.x;
    const int tid = threadIdx.x;
    const float* y = Y + (size_t)b * 3072;

    __shared__ float sA[256], sB[256];
    float s = 0.f, s2 = 0.f;
    for (int i = tid; i < 3072; i += 256) { float v = y[i]; s += v; s2 += v * v; }
    sA[tid] = s; sB[tid] = s2;
    __syncthreads();
    for (int o = 128; o > 0; o >>= 1) {
        if (tid < o) { sA[tid] += sA[tid + o]; sB[tid] += sB[tid + o]; }
        __syncthreads();
    }
    const float mean = sA[0] * (1.f / 3072.f);
    const float var  = sB[0] * (1.f / 3072.f) - mean * mean;
    const float rstd = rsqrtf(var + LN_EPS_);

    for (int i = tid; i < DETER_; i += 256) {
        float r = (y[i]        - mean) * rstd * ln_g[i]        + ln_b[i];
        float c = (y[i + 1024] - mean) * rstd * ln_g[i + 1024] + ln_b[i + 1024];
        float u = (y[i + 2048] - mean) * rstd * ln_g[i + 2048] + ln_b[i + 2048];
        r = sigmoid_f(r);
        c = tanhf(r * c);
        u = sigmoid_f(u - 1.0f);                  // UPDATE_BIAS = -1
        const float dn = u * c + (1.f - u) * deterC[(size_t)b * DETER_ + i];
        deterC[(size_t)b * DETER_ + i] = dn;      // carry
        Xobs[(size_t)b * 2560 + i] = (_Float16)dn;
        Xens[(size_t)b * DETER_ + i] = (_Float16)dn;
        dout[((size_t)b * T_ + t) * OUTW_ + 192 + i] = dn;
    }
}

// ---------------------------------------------------------------------------
// Sample: post/prior stoch from (mean, std, eps); post -> stoch carry.
// ---------------------------------------------------------------------------
__global__ void rssm_sample(float* __restrict__ dout,
                            const float* __restrict__ epsPost,
                            const float* __restrict__ epsPrior,
                            float* __restrict__ stochC, int t)
{
    const int idx = blockIdx.x * blockDim.x + threadIdx.x;
    if (idx >= B_ * STOCH_) return;
    const int b = idx / STOCH_, s = idx % STOCH_;
    const size_t base = ((size_t)b * T_ + t) * OUTW_;
    const size_t ein  = ((size_t)b * T_ + t) * STOCH_ + s;

    const float om = dout[base + 0 + s],  os = dout[base + 32 + s];
    const float post = om + os * epsPost[ein];
    dout[base + 64 + s] = post;
    stochC[idx] = post;

    const float pm = dout[base + 96 + s], ps = dout[base + 128 + s];
    dout[base + 160 + s] = pm + ps * epsPrior[ein];
}

// ---------------------------------------------------------------------------
extern "C" void kernel_launch(void* const* d_in, const int* in_sizes, int n_in,
                              void* d_out, int out_size, void* d_ws, size_t ws_size,
                              hipStream_t stream)
{
    const float* embed    = (const float*)d_in[0];
    const float* action   = (const float*)d_in[1];
    const float* epsPost  = (const float*)d_in[2];
    const float* epsPrior = (const float*)d_in[3];
    const unsigned char* isFirst = (const unsigned char*)d_in[4];
    const int*   ensIdx   = (const int*)d_in[5];
    const float* W_gru = (const float*)d_in[6];
    const float* b_gru = (const float*)d_in[7];
    const float* ln_g  = (const float*)d_in[8];
    const float* ln_b  = (const float*)d_in[9];
    const float* W_inp = (const float*)d_in[10];
    const float* b_inp = (const float*)d_in[11];
    const float* W_obs = (const float*)d_in[12];
    const float* b_obs = (const float*)d_in[13];
    const float* W_ens = (const float*)d_in[14];
    const float* b_ens = (const float*)d_in[15];
    const float* W_obsd = (const float*)d_in[16];
    const float* b_obsd = (const float*)d_in[17];
    const float* W_ensd = (const float*)d_in[18];
    const float* b_ensd = (const float*)d_in[19];
    float* dout = (float*)d_out;

    // ---- workspace carve-out (256B aligned) ----
    char* ws = (char*)d_ws;
    size_t off = 0;
    auto alloc = [&](size_t bytes) -> char* {
        char* p = ws + off;
        off += (bytes + 255) & ~(size_t)255;
        return p;
    };
    _Float16* WtInp  = (_Float16*)alloc((size_t)1024 * 64 * 2);
    _Float16* WtGru  = (_Float16*)alloc((size_t)3072 * 2048 * 2);
    _Float16* WtObs  = (_Float16*)alloc((size_t)1024 * 2560 * 2);
    _Float16* WtEns  = (_Float16*)alloc((size_t)ENS_ * 1024 * 1024 * 2);
    _Float16* WtEnsD = (_Float16*)alloc((size_t)ENS_ * 64 * 1024 * 2);
    _Float16* WtObsD = (_Float16*)alloc((size_t)64 * 1024 * 2);
    _Float16* Xinp   = (_Float16*)alloc((size_t)B_ * 64 * 2);
    _Float16* Xgru   = (_Float16*)alloc((size_t)B_ * 2048 * 2);
    float*    Ygru   = (float*)   alloc((size_t)B_ * 3072 * 4);
    _Float16* Xobs   = (_Float16*)alloc((size_t)B_ * 2560 * 2);
    _Float16* Xens   = (_Float16*)alloc((size_t)B_ * 1024 * 2);
    _Float16* Hens   = (_Float16*)alloc((size_t)B_ * 1024 * 2);
    _Float16* Hobs   = (_Float16*)alloc((size_t)B_ * 1024 * 2);
    float*    deterC = (float*)   alloc((size_t)B_ * DETER_ * 4);
    float*    stochC = (float*)   alloc((size_t)B_ * STOCH_ * 4);

    // ---- one-time weight convert/transpose to f16 [N][Kpad] ----
    auto conv = [&](const float* W, _Float16* Wt, int K, int N, int Kpad) {
        long total = (long)N * Kpad;
        int blocks = (int)((total + 255) / 256);
        rssm_conv_w<<<blocks, 256, 0, stream>>>(W, Wt, K, N, Kpad);
    };
    conv(W_inp, WtInp, 38, 1024, 64);            // pad K 38 -> 64
    conv(W_gru, WtGru, 2048, 3072, 2048);
    conv(W_obs, WtObs, 2560, 1024, 2560);
    for (int e = 0; e < ENS_; ++e) {
        conv(W_ens  + (size_t)e * 1024 * 1024, WtEns  + (size_t)e * 1024 * 1024, 1024, 1024, 1024);
        conv(W_ensd + (size_t)e * 1024 * 64,   WtEnsD + (size_t)e * 64 * 1024,   1024, 64,   1024);
    }
    conv(W_obsd, WtObsD, 1024, 64, 1024);

    const dim3 gB(128);
    // ---- sequential scan over T ----
    for (int t = 0; t < T_; ++t) {
        rssm_prep<<<B_, 256, 0, stream>>>(embed, action, isFirst, stochC, deterC,
                                          Xinp, Xgru, Xobs, t);
        // img_inp: [256x64]@[64x1024] -> elu -> f16 (left half of GRU input)
        rssm_gemm<1><<<dim3(4, 16), gB, 0, stream>>>(
            Xinp, 64, WtInp, 64, b_inp, 64,
            nullptr, Xgru, 2048, nullptr, t, 0L, 0, nullptr, 0, 0);
        // GRU: [256x2048]@[2048x3072] + b -> raw f32
        rssm_gemm<0><<<dim3(4, 48), gB, 0, stream>>>(
            Xgru, 2048, WtGru, 2048, b_gru, 2048,
            Ygru, nullptr, 3072, nullptr, t, 0L, 0, nullptr, 0, 0);
        rssm_gate<<<B_, 256, 0, stream>>>(Ygru, ln_g, ln_b, deterC, Xobs, Xens, dout, t);
        // ensemble head: deter_new @ W_ens[idx] -> elu -> f16
        rssm_gemm<1><<<dim3(4, 16), gB, 0, stream>>>(
            Xens, 1024, WtEns, 1024, b_ens, 1024,
            nullptr, Hens, 1024, ensIdx, t, (long)1024 * 1024, 1024, nullptr, 0, 0);
        // ensemble dist: h @ W_ens_dist[idx] -> pm / softplus ps
        rssm_gemm<2><<<dim3(4, 1), gB, 0, stream>>>(
            Hens, 1024, WtEnsD, 1024, b_ensd, 1024,
            nullptr, nullptr, 0, ensIdx, t, (long)64 * 1024, 64, dout, 96, 128);
        // obs: [deter_new | embed] @ W_obs -> elu -> f16
        rssm_gemm<1><<<dim3(4, 16), gB, 0, stream>>>(
            Xobs, 2560, WtObs, 2560, b_obs, 2560,
            nullptr, Hobs, 1024, nullptr, t, 0L, 0, nullptr, 0, 0);
        // obs dist: xo @ W_obs_dist -> om / softplus os
        rssm_gemm<2><<<dim3(4, 1), gB, 0, stream>>>(
            Hobs, 1024, WtObsD, 1024, b_obsd, 1024,
            nullptr, nullptr, 0, nullptr, t, 0L, 0, dout, 0, 32);
        rssm_sample<<<(B_ * STOCH_ + 255) / 256, 256, 0, stream>>>(
            dout, epsPost, epsPrior, stochC, t);
    }
    (void)in_sizes; (void)n_in; (void)out_size; (void)ws_size;
}
